// HybridReasoningModule_77378130804847
// MI455X (gfx1250) — matrix operand order repaired and checked by reading, not compiled, not run on hardware
//
#include <hip/hip_runtime.h>
#include <math.h>

// ---------------- problem constants ----------------
#define BB 8
#define SS 1024
#define EE 1024
#define HH 16
#define DD 64
#define LL 2
#define MM (BB * SS)          // 8192 rows in all big GEMMs

// ---------------- WMMA vector types ----------------
typedef __attribute__((ext_vector_type(16))) __bf16 v16bf;
typedef __attribute__((ext_vector_type(2)))  __bf16 v2bf;
typedef __attribute__((ext_vector_type(8)))  float  v8f;

union BF16Frag { v16bf v; unsigned u[8]; };
union F32Acc   { v8f  v; float    f[8]; };

// Native packed fp32->bf16 (1 VALU op) with compile-safe fallback.
__device__ inline unsigned pack2(float lo, float hi) {
#if defined(__AMDGCN__) && __has_builtin(__builtin_amdgcn_cvt_pk_bf16_f32)
    v2bf p = __builtin_amdgcn_cvt_pk_bf16_f32(lo, hi);
    return __builtin_bit_cast(unsigned, p);
#else
    v2bf p; p.x = (__bf16)lo; p.y = (__bf16)hi;
    return __builtin_bit_cast(unsigned, p);
#endif
}
__device__ inline unsigned short bf16_1(float x) {
    __bf16 h = (__bf16)x;
    return __builtin_bit_cast(unsigned short, h);
}
__device__ inline float gelu_exact(float x) {
    return 0.5f * x * (1.0f + erff(x * 0.70710678118654752f));
}
__device__ inline v8f wmma_bf16(v16bf a, v16bf b, v8f c) {
    return __builtin_amdgcn_wmma_f32_16x16x32_bf16(false, a, false, b, (short)0, c, false, false);
}

// Build A fragment (16x32 bf16) from LDS tile [rows][stride ushort].
// Lanes 0-15 row M=lane, lanes 16-31 same rows; VGPR0-3: K=(2i,2i+1)+16*half,
// VGPR4-7: K=8+(2i,2i+1)+16*half  (contiguous K, matches lds_bfrag).
__device__ inline v16bf lds_afrag(const unsigned short* s, int rowbase, int lane, int stride) {
    BF16Frag f;
    const int r  = rowbase + (lane & 15);
    const int h8 = (lane >> 4) << 3;
    #pragma unroll
    for (int i = 0; i < 4; ++i)
        f.u[i]     = *(const unsigned*)(s + r * stride + (2 * i + 2 * h8));
    #pragma unroll
    for (int i = 0; i < 4; ++i)
        f.u[4 + i] = *(const unsigned*)(s + r * stride + (8 + 2 * i + 2 * h8));
    return f.v;
}
// Build B fragment (32x16 bf16): LDS rows = N, cols = K contraction.
// Lanes 0-15: K=0-15 (2/VGPR), lanes 16-31: K=16-31.
__device__ inline v16bf lds_bfrag(const unsigned short* s, int nbase, int lane, int stride) {
    BF16Frag f;
    const int n  = nbase + (lane & 15);
    const int cb = (lane >> 4) << 4;
    #pragma unroll
    for (int i = 0; i < 8; ++i)
        f.u[i] = *(const unsigned*)(s + n * stride + (cb + 2 * i));
    return f.v;
}

// =====================================================================
// GEMM: C[M,N] = act( A[M,K] @ W[N,K]^T )
// BM=64, BN=128, BK=32, 256 threads = 8 waves; wave -> 16x64 of C
// (1 A-frag reused over 4 B-frags = 4 WMMA / k-tile / wave).
// Software-pipelined: global loads for k+32 issued while k is consumed.
// ACT: 0 = none, 1 = exact GELU
// =====================================================================
#define LDS_STRIDE 36   // 32 + 4 pad (ushorts) -> conflict-free pair reads
template <int ACT>
__global__ __launch_bounds__(256) void wmma_gemm_nt(const float* __restrict__ A,
                                                    const float* __restrict__ W,
                                                    float* __restrict__ C,
                                                    int M, int N, int K) {
    __shared__ unsigned short a_s[64  * LDS_STRIDE];
    __shared__ unsigned short w_s[128 * LDS_STRIDE];

    const int tid  = threadIdx.x;
    const int lane = tid & 31;
    const int wave = tid >> 5;
    const int wr = wave & 3;          // 4 row sub-tiles of 16
    const int wc = wave >> 2;         // 2 col sub-tiles of 64
    const int rowbase = blockIdx.x * 64;
    const int colbase = blockIdx.y * 128;

    F32Acc acc[4];
    #pragma unroll
    for (int c = 0; c < 4; ++c)
        #pragma unroll
        for (int r = 0; r < 8; ++r) acc[c].f[r] = 0.f;

    // A staging: 64x32 -> 8 floats/thread
    const int lrA = tid >> 2, lkA = (tid & 3) << 3;
    // W staging: 128x32 -> 16 floats/thread (two 8-float chunks)
    const int lrW = tid >> 2, lkW = (tid & 3) << 3;   // rows 0..63 and +64

    const float* gA  = A + (size_t)(rowbase + lrA) * K + lkA;
    const float* gW0 = W + (size_t)(colbase + lrW) * K + lkW;
    const float* gW1 = W + (size_t)(colbase + lrW + 64) * K + lkW;

    float4 a0, a1, p0, p1, q0, q1;
    a0 = ((const float4*)gA)[0];  a1 = ((const float4*)gA)[1];
    p0 = ((const float4*)gW0)[0]; p1 = ((const float4*)gW0)[1];
    q0 = ((const float4*)gW1)[0]; q1 = ((const float4*)gW1)[1];

    for (int kt = 0; kt < K; kt += 32) {
        // ---- convert + store current tile to LDS (packed cvt) ----
        unsigned* pa  = (unsigned*)(a_s + lrA * LDS_STRIDE + lkA);
        unsigned* pw0 = (unsigned*)(w_s + lrW * LDS_STRIDE + lkW);
        unsigned* pw1 = (unsigned*)(w_s + (lrW + 64) * LDS_STRIDE + lkW);
        pa[0]  = pack2(a0.x, a0.y); pa[1]  = pack2(a0.z, a0.w);
        pa[2]  = pack2(a1.x, a1.y); pa[3]  = pack2(a1.z, a1.w);
        pw0[0] = pack2(p0.x, p0.y); pw0[1] = pack2(p0.z, p0.w);
        pw0[2] = pack2(p1.x, p1.y); pw0[3] = pack2(p1.z, p1.w);
        pw1[0] = pack2(q0.x, q0.y); pw1[1] = pack2(q0.z, q0.w);
        pw1[2] = pack2(q1.x, q1.y); pw1[3] = pack2(q1.z, q1.w);
        __syncthreads();

        // ---- issue next tile's global loads while WMMAs consume this one ----
        if (kt + 32 < K) {
            const float* nA  = gA  + kt + 32;
            const float* nW0 = gW0 + kt + 32;
            const float* nW1 = gW1 + kt + 32;
            a0 = ((const float4*)nA)[0];  a1 = ((const float4*)nA)[1];
            p0 = ((const float4*)nW0)[0]; p1 = ((const float4*)nW0)[1];
            q0 = ((const float4*)nW1)[0]; q1 = ((const float4*)nW1)[1];
            if (kt + 64 < K) {           // hint L2 for the tile after next
                __builtin_prefetch(gA  + kt + 64, 0, 1);
                __builtin_prefetch(gW0 + kt + 64, 0, 1);
                __builtin_prefetch(gW1 + kt + 64, 0, 1);
            }
        }

        v16bf af = lds_afrag(a_s, wr * 16, lane, LDS_STRIDE);
        #pragma unroll
        for (int c = 0; c < 4; ++c) {
            v16bf bfr = lds_bfrag(w_s, wc * 64 + c * 16, lane, LDS_STRIDE);
            acc[c].v = wmma_bf16(af, bfr, acc[c].v);
        }
        __syncthreads();
    }

    // ---- epilogue: C layout = VGPR r -> row r + 8*(lane/16), col = lane%16 ----
    const int m0 = rowbase + wr * 16 + ((lane >> 4) << 3);
    const int n0 = colbase + wc * 64 + (lane & 15);
    #pragma unroll
    for (int r = 0; r < 8; ++r) {
        #pragma unroll
        for (int c = 0; c < 4; ++c) {
            float v = acc[c].f[r];
            if (ACT == 1) v = gelu_exact(v);
            C[(size_t)(m0 + r) * N + n0 + c * 16] = v;
        }
    }
}

// =====================================================================
// Flash attention over packed qkv [B,S,3E].  One wave = 16 query rows of
// one (b,h).  Block = 8 waves; grid = (S/128, B*H).  Key block = 32.
// =====================================================================
#define P_STRIDE 34
__global__ __launch_bounds__(256) void flash_attn_kernel(const float* __restrict__ qkv,
                                                         float* __restrict__ out) {
    __shared__ unsigned short p_s[8][16 * P_STRIDE];   // per-wave P bounce (C->A relayout)

    const int lane = threadIdx.x & 31;
    const int wave = threadIdx.x >> 5;
    const int b = blockIdx.y >> 4;
    const int h = blockIdx.y & 15;
    const int qbase = blockIdx.x * 128 + wave * 16;

    const size_t base = (size_t)b * SS * 3 * EE + (size_t)h * DD;
    const float* qp = qkv + base;
    const float* kp = qkv + base + EE;
    const float* vp = qkv + base + 2 * EE;

    const int row = lane & 15;
    const int h8  = (lane >> 4) << 3;
    const int cb  = (lane >> 4) << 4;
    const float scale = 0.125f;      // 1/sqrt(64), folded into Q

    // ---- Q fragments (2 chunks of K=32 over D=64), scaled, bf16 ----
    BF16Frag aq[2];
    {
        const float* qrow = qp + (size_t)(qbase + row) * (3 * EE);
        #pragma unroll
        for (int c = 0; c < 2; ++c) {
            #pragma unroll
            for (int i = 0; i < 8; ++i) {
                int d = c * 32 + cb + 2 * i;
                float2 q2 = *(const float2*)(qrow + d);
                aq[c].u[i] = pack2(q2.x * scale, q2.y * scale);
            }
        }
    }

    float mmax[8], lsum[8];
    F32Acc o[4];
    #pragma unroll
    for (int r = 0; r < 8; ++r) { mmax[r] = -1e30f; lsum[r] = 0.f; }
    #pragma unroll
    for (int cc = 0; cc < 4; ++cc)
        #pragma unroll
        for (int r = 0; r < 8; ++r) o[cc].f[r] = 0.f;

    for (int kb = 0; kb < SS; kb += 32) {
        // ---- scores S = Q K^T for 32 keys (two 16-wide column tiles) ----
        F32Acc s0, s1;
        #pragma unroll
        for (int r = 0; r < 8; ++r) { s0.f[r] = 0.f; s1.f[r] = 0.f; }
        #pragma unroll
        for (int c = 0; c < 2; ++c) {
            BF16Frag bk0, bk1;
            const float* k0 = kp + (size_t)(kb + row) * (3 * EE);
            const float* k1 = kp + (size_t)(kb + 16 + row) * (3 * EE);
            #pragma unroll
            for (int i = 0; i < 8; ++i) {
                int d = c * 32 + cb + 2 * i;
                float2 a2 = *(const float2*)(k0 + d);
                float2 b2 = *(const float2*)(k1 + d);
                bk0.u[i] = pack2(a2.x, a2.y);
                bk1.u[i] = pack2(b2.x, b2.y);
            }
            s0.v = wmma_bf16(aq[c].v, bk0.v, s0.v);
            s1.v = wmma_bf16(aq[c].v, bk1.v, s1.v);
        }
        // ---- online softmax (row stats across 16 lanes of the half-wave) ----
        #pragma unroll
        for (int r = 0; r < 8; ++r) {
            float mx = fmaxf(s0.f[r], s1.f[r]);
            #pragma unroll
            for (int m = 8; m >= 1; m >>= 1) mx = fmaxf(mx, __shfl_xor(mx, m, 16));
            float mnew = fmaxf(mmax[r], mx);
            float p0 = __expf(s0.f[r] - mnew);
            float p1 = __expf(s1.f[r] - mnew);
            float rs = p0 + p1;
            #pragma unroll
            for (int m = 8; m >= 1; m >>= 1) rs += __shfl_xor(rs, m, 16);
            float corr = __expf(mmax[r] - mnew);
            lsum[r] = lsum[r] * corr + rs;
            mmax[r] = mnew;
            #pragma unroll
            for (int cc = 0; cc < 4; ++cc) o[cc].f[r] *= corr;
            s0.f[r] = p0; s1.f[r] = p1;
        }
        // ---- bounce P through LDS: C layout -> A-fragment layout ----
        __syncthreads();
        unsigned short* ps = &p_s[wave][0];
        #pragma unroll
        for (int r = 0; r < 8; ++r) {
            int m = r + h8;
            ps[m * P_STRIDE + (lane & 15)]      = bf16_1(s0.f[r]);
            ps[m * P_STRIDE + 16 + (lane & 15)] = bf16_1(s1.f[r]);
        }
        __syncthreads();
        BF16Frag ap;
        #pragma unroll
        for (int i = 0; i < 8; ++i) {
            int c = cb + 2 * i;
            ap.u[i] = *(const unsigned*)(ps + row * P_STRIDE + c);
        }
        // ---- O += P @ V  (4 output chunks of 16 along D) ----
        #pragma unroll
        for (int cc = 0; cc < 4; ++cc) {
            BF16Frag bv;
            int d = cc * 16 + (lane & 15);
            #pragma unroll
            for (int i = 0; i < 8; ++i) {
                int key = kb + cb + 2 * i;
                bv.u[i] = pack2(vp[(size_t)key * (3 * EE) + d],
                                vp[(size_t)(key + 1) * (3 * EE) + d]);
            }
            o[cc].v = wmma_bf16(ap.v, bv.v, o[cc].v);
        }
    }

    // ---- write O / l  to attn output [B,S,E] ----
    #pragma unroll
    for (int r = 0; r < 8; ++r) {
        int s = qbase + r + h8;
        float inv = 1.0f / lsum[r];
        #pragma unroll
        for (int cc = 0; cc < 4; ++cc)
            out[(size_t)b * SS * EE + (size_t)s * EE + h * DD + cc * 16 + (lane & 15)] =
                o[cc].f[r] * inv;
    }
}

// =====================================================================
// Fused residual-add + LayerNorm: out = LN(x + res) * g + b, row = 1024
// =====================================================================
__global__ __launch_bounds__(256) void add_layernorm(const float* __restrict__ x,
                                                     const float* __restrict__ res,
                                                     const float* __restrict__ g,
                                                     const float* __restrict__ bta,
                                                     float* __restrict__ out) {
    __shared__ float s_sum[8], s_sq[8];
    const int tid = threadIdx.x;
    const size_t ro = (size_t)blockIdx.x * EE;
    float v[4]; float sum = 0.f, sq = 0.f;
    #pragma unroll
    for (int i = 0; i < 4; ++i) {
        int c = tid + i * 256;
        float a = x[ro + c] + res[ro + c];
        v[i] = a; sum += a; sq += a * a;
    }
    #pragma unroll
    for (int m = 16; m >= 1; m >>= 1) { sum += __shfl_xor(sum, m, 32); sq += __shfl_xor(sq, m, 32); }
    if ((tid & 31) == 0) { s_sum[tid >> 5] = sum; s_sq[tid >> 5] = sq; }
    __syncthreads();
    if (tid < 32) {
        float a  = (tid < 8) ? s_sum[tid] : 0.f;
        float c2 = (tid < 8) ? s_sq[tid]  : 0.f;
        #pragma unroll
        for (int m = 4; m >= 1; m >>= 1) { a += __shfl_xor(a, m, 32); c2 += __shfl_xor(c2, m, 32); }
        if (tid == 0) { s_sum[0] = a; s_sq[0] = c2; }
    }
    __syncthreads();
    const float mu   = s_sum[0] * (1.f / EE);
    const float var  = s_sq[0] * (1.f / EE) - mu * mu;
    const float rstd = rsqrtf(var + 1e-5f);
    #pragma unroll
    for (int i = 0; i < 4; ++i) {
        int c = tid + i * 256;
        out[ro + c] = (v[i] - mu) * rstd * g[c] + bta[c];
    }
}

// ---------------- mean-pool over S ----------------
__global__ void pool_mean(const float* __restrict__ x, float* __restrict__ pooled) {
    const int e = blockIdx.x * 256 + threadIdx.x;
    const int b = blockIdx.y;
    float s = 0.f;
    for (int t = 0; t < SS; ++t) s += x[(size_t)b * SS * EE + (size_t)t * EE + e];
    pooled[b * EE + e] = s * (1.0f / SS);
}

// ---------------- complexity estimator MLP + sigmoid (tiny) ----------------
__global__ __launch_bounds__(256) void ce_router(const float* __restrict__ pooled,
                                                 const float* __restrict__ w1,
                                                 const float* __restrict__ b1,
                                                 const float* __restrict__ w2,
                                                 const float* __restrict__ b2,
                                                 float* __restrict__ probs,
                                                 float* __restrict__ out_tail) {
    __shared__ float hsh[BB][256];
    __shared__ float psh[BB];
    const int j = threadIdx.x;    // 0..255 = E/4 hidden units
    for (int b = 0; b < BB; ++b) {
        float acc = b1[j];
        const float* pb = pooled + b * EE;
        const float* wj = w1 + (size_t)j * EE;
        for (int k = 0; k < EE; ++k) acc += pb[k] * wj[k];
        hsh[b][j] = fmaxf(acc, 0.f) * w2[j];
    }
    __syncthreads();
    if (j < BB) {
        float s = b2[0];
        for (int k = 0; k < 256; ++k) s += hsh[j][k];
        float p = 1.f / (1.f + __expf(-s));
        psh[j] = p;
        probs[j] = p;
        out_tail[j] = p;                                // probs
        out_tail[BB + j] = (p > 0.5f) ? 1.f : 0.f;      // mode_f
    }
    __syncthreads();
    if (j == 0) {
        float ms = 0.f;
        for (int b = 0; b < BB; ++b) ms += (psh[b] > 0.5f) ? 1.f : 0.f;
        out_tail[2 * BB] = ms * (1.0f / BB);            // mode mean
    }
}

// ---------------- per-batch routed select (b in blockIdx.y, no int div) ----
__global__ void select_out(const float* __restrict__ t, const float* __restrict__ fast,
                           const float* __restrict__ probs, float* __restrict__ out) {
    const int b = blockIdx.y;
    size_t i = (size_t)b * SS * EE + (size_t)blockIdx.x * 256 + threadIdx.x;
    out[i] = (probs[b] > 0.5f) ? t[i] : fast[i];
}

// =====================================================================
extern "C" void kernel_launch(void* const* d_in, const int* in_sizes, int n_in,
                              void* d_out, int out_size, void* d_ws, size_t ws_size,
                              hipStream_t stream) {
    (void)in_sizes; (void)n_in; (void)out_size; (void)ws_size;
    const float* x      = (const float*)d_in[0];
    const float* w_in   = (const float*)d_in[1];   // [L,3E,E]
    const float* w_out  = (const float*)d_in[2];   // [L,E,E]
    const float* ffn_w1 = (const float*)d_in[3];   // [L,4E,E]
    const float* ffn_w2 = (const float*)d_in[4];   // [L,E,4E]
    const float* ln1_g  = (const float*)d_in[5];
    const float* ln1_b  = (const float*)d_in[6];
    const float* ln2_g  = (const float*)d_in[7];
    const float* ln2_b  = (const float*)d_in[8];
    const float* ce_w1  = (const float*)d_in[9];
    const float* ce_b1  = (const float*)d_in[10];
    const float* ce_w2  = (const float*)d_in[11];
    const float* ce_b2  = (const float*)d_in[12];
    const float* fast_w = (const float*)d_in[13];
    float* outf = (float*)d_out;

    const size_t SZ_E = (size_t)BB * SS * EE;   // 8,388,608 floats
    float* ws     = (float*)d_ws;
    float* t      = ws;                          // [B,S,E]
    float* bufA   = t    + SZ_E;                 // [B,S,4E]  (qkv / ffn hidden)
    float* bufB   = bufA + 4 * SZ_E;             // [B,S,E]   (attn out / ln1 out)
    float* bufC   = bufB + SZ_E;                 // [B,S,E]   (proj / ffn2 out)
    float* bufD   = bufC + SZ_E;                 // [B,S,E]   (fast path)
    float* pooled = bufD + SZ_E;                 // [B,E]
    float* probs  = pooled + BB * EE;            // [B]

    const dim3 blk(256);
    const dim3 gQKV(MM / 64, (3 * EE) / 128);
    const dim3 gE  (MM / 64, EE / 128);
    const dim3 g4E (MM / 64, (4 * EE) / 128);
    const dim3 gFA (SS / 128, BB * HH);

    // ---- thinking path: L transformer layers ----
    for (int i = 0; i < LL; ++i) {
        const float* cur = (i == 0) ? x : t;
        // qkv = cur @ w_in^T
        wmma_gemm_nt<0><<<gQKV, blk, 0, stream>>>(cur, w_in + (size_t)i * 3 * EE * EE,
                                                  bufA, MM, 3 * EE, EE);
        // attn = flash(qkv)
        flash_attn_kernel<<<gFA, blk, 0, stream>>>(bufA, bufB);
        // proj = attn @ w_out^T
        wmma_gemm_nt<0><<<gE, blk, 0, stream>>>(bufB, w_out + (size_t)i * EE * EE,
                                                bufC, MM, EE, EE);
        // x1 = LN1(cur + proj)
        add_layernorm<<<MM, blk, 0, stream>>>(cur, bufC, ln1_g + i * EE, ln1_b + i * EE, bufB);
        // h = gelu(x1 @ ffn1^T)
        wmma_gemm_nt<1><<<g4E, blk, 0, stream>>>(bufB, ffn_w1 + (size_t)i * 4 * EE * EE,
                                                 bufA, MM, 4 * EE, EE);
        // f2 = h @ ffn2^T
        wmma_gemm_nt<0><<<gE, blk, 0, stream>>>(bufA, ffn_w2 + (size_t)i * EE * 4 * EE,
                                                bufC, MM, EE, 4 * EE);
        // t = LN2(x1 + f2)
        add_layernorm<<<MM, blk, 0, stream>>>(bufB, bufC, ln2_g + i * EE, ln2_b + i * EE, t);
    }

    // ---- fast path: gelu(x @ fast_w^T) ----
    wmma_gemm_nt<1><<<gE, blk, 0, stream>>>(x, fast_w, bufD, MM, EE, EE);

    // ---- router ----
    pool_mean<<<dim3(EE / 256, BB), blk, 0, stream>>>(x, pooled);
    ce_router<<<1, blk, 0, stream>>>(pooled, ce_w1, ce_b1, ce_w2, ce_b2,
                                     probs, outf + SZ_E);

    // ---- routed select ----
    select_out<<<dim3((unsigned)(SS * EE / 256), BB), blk, 0, stream>>>(t, bufD, probs, outf);
}
